// GatedDeltaNet_42236708389006
// MI455X (gfx1250) — compile-verified, matrix operand order checked by reading, not run on hardware
//
#include <hip/hip_runtime.h>
#include <cstdint>
#include <cstddef>

// ---------------------------------------------------------------------------
// GatedDeltaNet forward for MI455X (gfx1250, wave32, WMMA bf16 16x16x32).
// B=2 T=2048 HID=2048 H=32 KVH=8 DK=DV=128.
// ---------------------------------------------------------------------------

typedef __bf16 bf16_t;
typedef __attribute__((ext_vector_type(16))) __bf16 v16bf;
typedef __attribute__((ext_vector_type(8)))  __bf16 v8bf;
typedef __attribute__((ext_vector_type(4)))  __bf16 v4bf;
typedef __attribute__((ext_vector_type(8)))  float  v8f;

#define WMMA_BF16(a, b, c) \
  __builtin_amdgcn_wmma_f32_16x16x32_bf16(false, (a), false, (b), (short)0, (c), false, false)

#if __has_builtin(__builtin_amdgcn_sched_barrier)
#define SCHED_FENCE() __builtin_amdgcn_sched_barrier(0)
#else
#define SCHED_FENCE()
#endif

static constexpr int Bb  = 2;
static constexpr int Tt  = 2048;
static constexpr int HID = 2048;
static constexpr int Hh  = 32;
static constexpr int KVH = 8;
static constexpr int DK  = 128;
static constexpr int DV  = 128;

__device__ __forceinline__ v8f zero8() {
  v8f z;
#pragma unroll
  for (int i = 0; i < 8; ++i) z[i] = 0.f;
  return z;
}

// A-fragment (16x32 bf16, row-major source, K contiguous): lane m=lane&15,
// half=lane>>4; f[j]=K(half*8+j), f[8+j]=K(16+half*8+j).  Two 16B LDS reads.
__device__ __forceinline__ v16bf frag_row_bf16(const bf16_t* base, int ld, int m0, int k0, int lane) {
  const int half = (lane >> 4) & 1, m = lane & 15;
  const bf16_t* p = base + (size_t)(m0 + m) * ld + k0 + half * 8;
  v8bf lo = *(const v8bf*)p;
  v8bf hi = *(const v8bf*)(p + 16);
  v16bf f;
#pragma unroll
  for (int j = 0; j < 8; ++j) { f[j] = lo[j]; f[8 + j] = hi[j]; }
  return f;
}

// Same pattern from an f32 row-major LDS source with on-the-fly bf16 convert.
// Also serves as a transposed-B loader (B = Src^T) since the B-fragment lane
// layout mirrors A's.
__device__ __forceinline__ v16bf frag_row_f32(const float* base, int ld, int m0, int k0, int lane) {
  const int half = (lane >> 4) & 1, m = lane & 15;
  const float* p = base + (size_t)(m0 + m) * ld + k0 + half * 8;
  v16bf f;
#pragma unroll
  for (int j = 0; j < 8; ++j) { f[j] = (bf16_t)p[j]; f[8 + j] = (bf16_t)p[16 + j]; }
  return f;
}

// Column fragment: element(x, k) = base[k*ld + x0 + x] (optional per-k scale).
__device__ __forceinline__ v16bf frag_col_f32(const float* base, int ld, int x0, int k0, int lane,
                                              const float* kscale) {
  const int half = (lane >> 4) & 1, x = lane & 15;
  v16bf f;
#pragma unroll
  for (int j = 0; j < 8; ++j) {
    const int k1 = k0 + half * 8 + j;
    const int k2 = k1 + 16;
    float a = base[(size_t)k1 * ld + x0 + x];
    float b = base[(size_t)k2 * ld + x0 + x];
    if (kscale) { a *= kscale[k1]; b *= kscale[k2]; }
    f[j] = (bf16_t)a;
    f[8 + j] = (bf16_t)b;
  }
  return f;
}

// C/D 16x16 f32 tile store: VGPR r -> row m0 + (lane>=16 ? 8 : 0) + r, col lane&15.
__device__ __forceinline__ void store_tile_f32(float* base, int ld, int m0, int n0, v8f acc, int lane) {
  const int half = (lane >> 4) & 1, nn = lane & 15;
#pragma unroll
  for (int r = 0; r < 8; ++r) base[(size_t)(m0 + half * 8 + r) * ld + n0 + nn] = acc[r];
}

// ---------------------------------------------------------------------------
// fp32 -> bf16 conversion (row-major copy, for activations)
// ---------------------------------------------------------------------------
__global__ __launch_bounds__(256) void cvt_bf16_kernel(const float* __restrict__ src,
                                                       bf16_t* __restrict__ dst, size_t n) {
  size_t stride = (size_t)gridDim.x * blockDim.x;
  for (size_t i = (size_t)blockIdx.x * blockDim.x + threadIdx.x; i < n; i += stride)
    dst[i] = (bf16_t)src[i];
}

// ---------------------------------------------------------------------------
// fp32 [K][N] -> bf16 [N][K] transpose-convert (32x32 LDS tiles, coalesced).
// ---------------------------------------------------------------------------
__global__ __launch_bounds__(256) void cvt_transpose_kernel(const float* __restrict__ src,
                                                            bf16_t* __restrict__ dst,
                                                            int K, int N) {
  __shared__ float tile[32][33];
  const int kb = blockIdx.y * 32, nb = blockIdx.x * 32;
  const int tx = threadIdx.x & 31, ty = threadIdx.x >> 5;  // ty: 0..7
#pragma unroll
  for (int i = 0; i < 32; i += 8)
    tile[ty + i][tx] = src[(size_t)(kb + ty + i) * N + nb + tx];
  __syncthreads();
#pragma unroll
  for (int i = 0; i < 32; i += 8)
    dst[(size_t)(nb + ty + i) * K + kb + tx] = (bf16_t)tile[tx][ty + i];
}

// ---------------------------------------------------------------------------
// Pack Wb[K][32] and Wa[K][32] into one zero-padded transposed bf16 [128][K]:
// rows 0..31 = Wb^T, rows 32..63 = Wa^T, rows 64..127 = 0.
// ---------------------------------------------------------------------------
__global__ __launch_bounds__(256) void pack_ba_kernel(const float* __restrict__ Wb,
                                                      const float* __restrict__ Wa,
                                                      bf16_t* __restrict__ dst, int K) {
  const int total = 128 * K;
  for (int idx = blockIdx.x * 256 + threadIdx.x; idx < total; idx += gridDim.x * 256) {
    const int n = idx / K, k = idx - n * K;
    float v = 0.f;
    if (n < 32)      v = Wb[(size_t)k * 32 + n];
    else if (n < 64) v = Wa[(size_t)k * 32 + (n - 32)];
    dst[idx] = (bf16_t)v;
  }
}

// ---------------------------------------------------------------------------
// Tiled bf16 GEMM: C[M,N] = A[M,K] @ Bt[N,K]^T, fp32 out.  Full tiles only
// (M,N multiples of 128; K multiple of 32).  128x128x32 tiles, 8 waves (4x2),
// each wave 32x64 = 2x4 WMMA tiles.  LDS double-buffered with register
// staging; a sched_barrier pins the next tile's global_load_b128 issue ahead
// of the WMMA block so the HBM/L2 latency is hidden under compute.
// ---------------------------------------------------------------------------
__global__ __launch_bounds__(256) void gemm_bf16_tn_kernel(const bf16_t* __restrict__ A,
                                                           const bf16_t* __restrict__ Bt,
                                                           float* __restrict__ C,
                                                           int M, int N, int K) {
  __shared__ __align__(16) bf16_t As[2][128 * 40];   // [m][k], ld 40 (16B-aligned rows)
  __shared__ __align__(16) bf16_t Bs[2][128 * 40];   // [n][k]

  const int tid = threadIdx.x, lane = tid & 31, wave = tid >> 5;
  const int wm = (wave >> 1) * 32, wn = (wave & 1) * 64;
  const size_t m_base = (size_t)blockIdx.y * 128, n_base = (size_t)blockIdx.x * 128;

  // 512 chunks of 8 bf16 per tile; thread handles chunks tid and tid+256.
  // chunk c: row = c>>2 (0..127), koff = (c&3)*8.
  const int r0 = tid >> 2, ko = (tid & 3) * 8;
  const bf16_t* Aptr = A + m_base * K;
  const bf16_t* Bptr = Bt + n_base * K;

  v8bf a0r, a1r, b0r, b1r;
  auto g_load = [&](int k0) {
    a0r = *(const v8bf*)(Aptr + (size_t)r0 * K + k0 + ko);
    a1r = *(const v8bf*)(Aptr + (size_t)(r0 + 64) * K + k0 + ko);
    b0r = *(const v8bf*)(Bptr + (size_t)r0 * K + k0 + ko);
    b1r = *(const v8bf*)(Bptr + (size_t)(r0 + 64) * K + k0 + ko);
  };
  auto s_store = [&](int buf) {
    *(v8bf*)(&As[buf][r0 * 40 + ko]) = a0r;
    *(v8bf*)(&As[buf][(r0 + 64) * 40 + ko]) = a1r;
    *(v8bf*)(&Bs[buf][r0 * 40 + ko]) = b0r;
    *(v8bf*)(&Bs[buf][(r0 + 64) * 40 + ko]) = b1r;
  };

  v8f acc[2][4];
#pragma unroll
  for (int i = 0; i < 2; ++i)
#pragma unroll
    for (int j = 0; j < 4; ++j) acc[i][j] = zero8();

  g_load(0);
  s_store(0);
  __syncthreads();

  int cur = 0;
  for (int k0 = 32; k0 < K; k0 += 32) {
    g_load(k0);      // issue next tile's global loads ...
    SCHED_FENCE();   // ... and keep them issued BEFORE the WMMA block below
    {
      v16bf a0 = frag_row_bf16(As[cur], 40, wm, 0, lane);
      v16bf a1 = frag_row_bf16(As[cur], 40, wm + 16, 0, lane);
#pragma unroll
      for (int j = 0; j < 4; ++j) {
        v16bf bf = frag_row_bf16(Bs[cur], 40, wn + j * 16, 0, lane);
        acc[0][j] = WMMA_BF16(a0, bf, acc[0][j]);
        acc[1][j] = WMMA_BF16(a1, bf, acc[1][j]);
      }
    }
    __syncthreads();
    s_store(cur ^ 1);
    __syncthreads();
    cur ^= 1;
  }
  {
    v16bf a0 = frag_row_bf16(As[cur], 40, wm, 0, lane);
    v16bf a1 = frag_row_bf16(As[cur], 40, wm + 16, 0, lane);
#pragma unroll
    for (int j = 0; j < 4; ++j) {
      v16bf bf = frag_row_bf16(Bs[cur], 40, wn + j * 16, 0, lane);
      acc[0][j] = WMMA_BF16(a0, bf, acc[0][j]);
      acc[1][j] = WMMA_BF16(a1, bf, acc[1][j]);
    }
  }

  const int half = (lane >> 4) & 1, nn = lane & 15;
#pragma unroll
  for (int i = 0; i < 2; ++i)
#pragma unroll
    for (int j = 0; j < 4; ++j)
#pragma unroll
      for (int r = 0; r < 8; ++r) {
        const size_t row = m_base + wm + i * 16 + half * 8 + r;
        const size_t col = n_base + wn + j * 16 + nn;
        C[row * N + col] = acc[i][j][r];
      }
}

// ---------------------------------------------------------------------------
// beta = 2*sigmoid(ba[:, h]); g = -exp(A_log[h]) * softplus(ba[:, 32+h] + dt_bias[h])
// ba is the packed [BT][128] output of the padded projection.
// ---------------------------------------------------------------------------
__global__ __launch_bounds__(256) void betag_kernel(const float* __restrict__ ba,
                                                    const float* __restrict__ A_log,
                                                    const float* __restrict__ dt_bias,
                                                    float* __restrict__ beta_o,
                                                    float* __restrict__ g_o, int n) {
  const int i = blockIdx.x * 256 + threadIdx.x;
  if (i >= n) return;
  const int row = i >> 5, h = i & (Hh - 1);
  const float braw = ba[(size_t)row * 128 + h];
  const float araw = ba[(size_t)row * 128 + 32 + h];
  beta_o[i] = 2.0f / (1.0f + __expf(-braw));
  const float x = araw + dt_bias[h];
  const float sp = (x > 15.0f) ? x : log1pf(__expf(x));
  g_o[i] = -__expf(A_log[h]) * sp;
}

// ---------------------------------------------------------------------------
// In-place L2-normalize rows of 128, then multiply by scale (1/sqrt(DK) for q).
// ---------------------------------------------------------------------------
__global__ __launch_bounds__(256) void l2norm_kernel(float* __restrict__ x, int nrows, float scale) {
  const int row = blockIdx.x * 8 + (threadIdx.x >> 5);
  if (row >= nrows) return;
  const int lane = threadIdx.x & 31;
  float4* p = (float4*)(x + (size_t)row * 128);
  float4 v = p[lane];
  float ss = v.x * v.x + v.y * v.y + v.z * v.z + v.w * v.w;
#pragma unroll
  for (int off = 16; off > 0; off >>= 1) ss += __shfl_xor(ss, off, 32);
  const float r = rsqrtf(ss + 1e-6f) * scale;
  v.x *= r; v.y *= r; v.z *= r; v.w *= r;
  p[lane] = v;
}

// ---------------------------------------------------------------------------
// Chunked gated delta rule. One workgroup (8 waves) per (b, h) chain; S (128x128
// f32) lives in LDS across all 32 chunks of 64 timesteps. Per chunk:
//   gc   = cumsum(g);  A[t,s] = beta_t e^{gc_t-gc_s} (k_t.k_s) (s<t)
//   U    = (I+A)^{-1} (beta o (V - e^{gc} o (K S_in)))          [fwd subst]
//   O    = e^{gc} o (Q S_in) + tril(e^{gc_t-gc_s} Q K^T) U
//   S    = e^{gc_C} S_in + (e^{gc_C-gc_s} k_s)^T U
// ---------------------------------------------------------------------------
__global__ __launch_bounds__(256) void delta_kernel(const float* __restrict__ qg,
                                                    const float* __restrict__ kg,
                                                    const float* __restrict__ vg,
                                                    const float* __restrict__ gg,
                                                    const float* __restrict__ bg,
                                                    float* __restrict__ og) {
  const int bh = blockIdx.x;
  const int b = bh / Hh, h = bh % Hh;
  const int kvh = h / (Hh / KVH);
  const int tid = threadIdx.x, lane = tid & 31, wave = tid >> 5;
  const int half = (lane >> 4) & 1, nn = lane & 15;

  __shared__ float S[128 * 132];
  __shared__ float Kf[64 * 132];
  __shared__ float Qs[64 * 128];
  __shared__ float Us[64 * 132];
  __shared__ float Ws[64 * 128];
  __shared__ float AM[64 * 68];
  __shared__ float gc[64];
  __shared__ float bb[64];
  __shared__ float ksc[64];

  for (int i = tid; i < 128 * 132; i += 256) S[i] = 0.f;
  __syncthreads();

  for (int t0 = 0; t0 < Tt; t0 += 64) {
    for (int idx = tid; idx < 64 * 128; idx += 256) {
      const int t = idx >> 7, d = idx & 127;
      const size_t qoff = ((((size_t)b * Tt + t0 + t) * Hh) + h) * DK + d;
      const size_t koff = ((((size_t)b * Tt + t0 + t) * KVH) + kvh) * DK + d;
      Qs[t * 128 + d] = qg[qoff];
      Kf[t * 132 + d] = kg[koff];
    }
    if (tid < 64) {
      const size_t off = (((size_t)b * Tt + t0 + tid) * Hh) + h;
      bb[tid] = bg[off];
      gc[tid] = gg[off];
    }
    __syncthreads();
    if (tid == 0) {
      float c = 0.f;
      for (int t = 0; t < 64; ++t) { c += gc[t]; gc[t] = c; }
    }
    __syncthreads();
    if (tid < 64) ksc[tid] = __expf(gc[63] - gc[tid]);

    // KS = K @ S_in
    for (int tile = wave; tile < 32; tile += 8) {
      const int tm = (tile >> 3) << 4, tn = (tile & 7) << 4;
      v8f acc = zero8();
#pragma unroll
      for (int kk = 0; kk < 128; kk += 32) {
        v16bf a = frag_row_f32(Kf, 132, tm, kk, lane);
        v16bf bf = frag_col_f32(S, 132, tn, kk, lane, nullptr);
        acc = WMMA_BF16(a, bf, acc);
      }
      store_tile_f32(Ws, 128, tm, tn, acc, lane);
    }
    // KK = K @ K^T
    for (int tile = wave; tile < 16; tile += 8) {
      const int tm = (tile >> 2) << 4, tn = (tile & 3) << 4;
      v8f acc = zero8();
#pragma unroll
      for (int kk = 0; kk < 128; kk += 32) {
        v16bf a = frag_row_f32(Kf, 132, tm, kk, lane);
        v16bf bf = frag_row_f32(Kf, 132, tn, kk, lane);
        acc = WMMA_BF16(a, bf, acc);
      }
      store_tile_f32(AM, 68, tm, tn, acc, lane);
    }
    __syncthreads();

    for (int idx = tid; idx < 64 * 128; idx += 256) {
      const int t = idx >> 7, d = idx & 127;
      const size_t voff = ((((size_t)b * Tt + t0 + t) * KVH) + kvh) * DV + d;
      Us[t * 132 + d] = bb[t] * (vg[voff] - __expf(gc[t]) * Ws[t * 128 + d]);
    }
    for (int idx = tid; idx < 64 * 64; idx += 256) {
      const int t = idx >> 6, s = idx & 63;
      const float v = AM[t * 68 + s];
      AM[t * 68 + s] = (s < t) ? bb[t] * __expf(gc[t] - gc[s]) * v : 0.f;
    }
    __syncthreads();

    if (tid < 128) {
      for (int t = 1; t < 64; ++t) {
        float a = 0.f;
        for (int s = 0; s < t; ++s) a += AM[t * 68 + s] * Us[s * 132 + tid];
        Us[t * 132 + tid] -= a;
      }
    }
    __syncthreads();

    // QS = Q @ S_in ; QK = Q @ K^T
    for (int tile = wave; tile < 32; tile += 8) {
      const int tm = (tile >> 3) << 4, tn = (tile & 7) << 4;
      v8f acc = zero8();
#pragma unroll
      for (int kk = 0; kk < 128; kk += 32) {
        v16bf a = frag_row_f32(Qs, 128, tm, kk, lane);
        v16bf bf = frag_col_f32(S, 132, tn, kk, lane, nullptr);
        acc = WMMA_BF16(a, bf, acc);
      }
      store_tile_f32(Ws, 128, tm, tn, acc, lane);
    }
    for (int tile = wave; tile < 16; tile += 8) {
      const int tm = (tile >> 2) << 4, tn = (tile & 3) << 4;
      v8f acc = zero8();
#pragma unroll
      for (int kk = 0; kk < 128; kk += 32) {
        v16bf a = frag_row_f32(Qs, 128, tm, kk, lane);
        v16bf bf = frag_row_f32(Kf, 132, tn, kk, lane);
        acc = WMMA_BF16(a, bf, acc);
      }
      store_tile_f32(AM, 68, tm, tn, acc, lane);
    }
    __syncthreads();
    for (int idx = tid; idx < 64 * 64; idx += 256) {
      const int t = idx >> 6, s = idx & 63;
      const float v = AM[t * 68 + s];
      AM[t * 68 + s] = (s <= t) ? __expf(gc[t] - gc[s]) * v : 0.f;
    }
    __syncthreads();

    // O = M @ U + e^{gc} o QS
    for (int tile = wave; tile < 32; tile += 8) {
      const int tm = (tile >> 3) << 4, tn = (tile & 7) << 4;
      v8f acc = zero8();
#pragma unroll
      for (int kk = 0; kk < 64; kk += 32) {
        v16bf a = frag_row_f32(AM, 68, tm, kk, lane);
        v16bf bf = frag_col_f32(Us, 132, tn, kk, lane, nullptr);
        acc = WMMA_BF16(a, bf, acc);
      }
#pragma unroll
      for (int r = 0; r < 8; ++r) {
        const int t = tm + half * 8 + r, d = tn + nn;
        const float val = acc[r] + __expf(gc[t]) * Ws[t * 128 + d];
        og[((((size_t)b * Tt + t0 + t) * Hh) + h) * DV + d] = val;
      }
    }
    __syncthreads();

    // S = e^{gc_C} S + Kd^T @ U
    const float expGC = __expf(gc[63]);
    for (int tile = wave; tile < 64; tile += 8) {
      const int tm = (tile >> 3) << 4, tn = (tile & 7) << 4;
      v8f acc;
#pragma unroll
      for (int r = 0; r < 8; ++r) acc[r] = expGC * S[(size_t)(tm + half * 8 + r) * 132 + tn + nn];
#pragma unroll
      for (int kk = 0; kk < 64; kk += 32) {
        v16bf a = frag_col_f32(Kf, 132, tm, kk, lane, ksc);
        v16bf bf = frag_col_f32(Us, 132, tn, kk, lane, nullptr);
        acc = WMMA_BF16(a, bf, acc);
      }
      store_tile_f32(S, 132, tm, tn, acc, lane);
    }
    __syncthreads();
  }
}

// ---------------------------------------------------------------------------
// o = rmsnorm(o) * w * gate * sigmoid(gate)  ->  bf16 rows for final GEMM
// ---------------------------------------------------------------------------
__global__ __launch_bounds__(256) void normgate_kernel(const float* __restrict__ o,
                                                       const float* __restrict__ gate,
                                                       const float* __restrict__ w,
                                                       bf16_t* __restrict__ ob, int nrows) {
  const int row = blockIdx.x * 8 + (threadIdx.x >> 5);
  if (row >= nrows) return;
  const int lane = threadIdx.x & 31;
  const float4 v = ((const float4*)(o + (size_t)row * 128))[lane];
  float ss = v.x * v.x + v.y * v.y + v.z * v.z + v.w * v.w;
#pragma unroll
  for (int off = 16; off > 0; off >>= 1) ss += __shfl_xor(ss, off, 32);
  const float r = rsqrtf(ss * (1.0f / 128.0f) + 1e-5f);
  const float4 wv = ((const float4*)w)[lane];
  const float4 gv = ((const float4*)(gate + (size_t)row * 128))[lane];
  v4bf out;
  out[0] = (bf16_t)(v.x * r * wv.x * gv.x / (1.0f + __expf(-gv.x)));
  out[1] = (bf16_t)(v.y * r * wv.y * gv.y / (1.0f + __expf(-gv.y)));
  out[2] = (bf16_t)(v.z * r * wv.z * gv.z / (1.0f + __expf(-gv.z)));
  out[3] = (bf16_t)(v.w * r * wv.w * gv.w / (1.0f + __expf(-gv.w)));
  ((v4bf*)(ob + (size_t)row * 128))[lane] = out;
}

// ---------------------------------------------------------------------------
extern "C" void kernel_launch(void* const* d_in, const int* in_sizes, int n_in,
                              void* d_out, int out_size, void* d_ws, size_t ws_size,
                              hipStream_t stream) {
  const float* h       = (const float*)d_in[0];
  const float* Wq      = (const float*)d_in[1];
  const float* Wk      = (const float*)d_in[2];
  const float* Wv      = (const float*)d_in[3];
  const float* Wb      = (const float*)d_in[4];
  const float* Wa      = (const float*)d_in[5];
  const float* Wg      = (const float*)d_in[6];
  const float* Wo      = (const float*)d_in[7];
  const float* A_log   = (const float*)d_in[8];
  const float* dt_bias = (const float*)d_in[9];
  const float* onw     = (const float*)d_in[10];
  float* out = (float*)d_out;
  (void)in_sizes; (void)n_in; (void)out_size; (void)ws_size;

  char* ws = (char*)d_ws;
  size_t off = 0;
  auto take = [&](size_t bytes) -> char* {
    char* p = ws + off;
    off += (bytes + 255) & ~(size_t)255;
    return p;
  };

  const size_t BT = (size_t)Bb * Tt;  // 4096 rows

  bf16_t* hb   = (bf16_t*)take(BT * HID * 2);
  bf16_t* WqT  = (bf16_t*)take((size_t)Hh * DK * HID * 2);   // [4096][2048]
  bf16_t* WkT  = (bf16_t*)take((size_t)KVH * DK * HID * 2);  // [1024][2048]
  bf16_t* WvT  = (bf16_t*)take((size_t)KVH * DV * HID * 2);  // [1024][2048]
  bf16_t* WbaT = (bf16_t*)take((size_t)128 * HID * 2);       // [128][2048] packed b|a|0
  bf16_t* WgT  = (bf16_t*)take((size_t)Hh * DV * HID * 2);   // [4096][2048]
  bf16_t* WoT  = (bf16_t*)take((size_t)HID * Hh * DV * 2);   // [2048][4096]
  float* q     = (float*)take(BT * Hh * DK * 4);
  float* k     = (float*)take(BT * KVH * DK * 4);
  float* v     = (float*)take(BT * KVH * DV * 4);
  float* ba    = (float*)take(BT * 128 * 4);
  float* beta  = (float*)take(BT * Hh * 4);
  float* g     = (float*)take(BT * Hh * 4);
  float* gate  = (float*)take(BT * Hh * DV * 4);
  float* o     = (float*)take(BT * Hh * DV * 4);
  bf16_t* ob   = (bf16_t*)take(BT * Hh * DV * 2);

  cvt_bf16_kernel<<<2048, 256, 0, stream>>>(h, hb, BT * HID);

  auto cvtT = [&](const float* s, bf16_t* d, int K_, int N_) {
    dim3 grid(N_ / 32, K_ / 32);
    cvt_transpose_kernel<<<grid, 256, 0, stream>>>(s, d, K_, N_);
  };
  cvtT(Wq, WqT, HID, Hh * DK);
  cvtT(Wk, WkT, HID, KVH * DK);
  cvtT(Wv, WvT, HID, KVH * DV);
  cvtT(Wg, WgT, HID, Hh * DV);
  cvtT(Wo, WoT, Hh * DV, HID);
  pack_ba_kernel<<<512, 256, 0, stream>>>(Wb, Wa, WbaT, HID);

  auto gemm = [&](const bf16_t* A, const bf16_t* Bt, float* C, int M, int N, int K) {
    dim3 grid(N / 128, M / 128);
    gemm_bf16_tn_kernel<<<grid, 256, 0, stream>>>(A, Bt, C, M, N, K);
  };
  gemm(hb, WqT, q,    (int)BT, Hh * DK,  HID);
  gemm(hb, WkT, k,    (int)BT, KVH * DK, HID);
  gemm(hb, WvT, v,    (int)BT, KVH * DV, HID);
  gemm(hb, WbaT, ba,  (int)BT, 128,      HID);
  gemm(hb, WgT, gate, (int)BT, Hh * DV,  HID);

  const int nBH = (int)(BT * Hh);
  betag_kernel<<<(nBH + 255) / 256, 256, 0, stream>>>(ba, A_log, dt_bias, beta, g, nBH);
  l2norm_kernel<<<(nBH + 7) / 8, 256, 0, stream>>>(q, nBH, 0.08838834764831845f);  // 1/sqrt(128)
  l2norm_kernel<<<((int)(BT * KVH) + 7) / 8, 256, 0, stream>>>(k, (int)(BT * KVH), 1.0f);

  delta_kernel<<<Bb * Hh, 256, 0, stream>>>(q, k, v, g, beta, o);

  normgate_kernel<<<nBH / 8, 256, 0, stream>>>(o, gate, onw, ob, nBH);
  gemm(ob, WoT, out, (int)BT, HID, Hh * DV);
}
